// SoftmaxAttention_1735166787667
// MI455X (gfx1250) — matrix operand order repaired and checked by reading, not compiled, not run on hardware
//
#include <hip/hip_runtime.h>

typedef __attribute__((ext_vector_type(16))) _Float16 v16h;
typedef __attribute__((ext_vector_type(8)))  float    v8f;

constexpr int S_LEN = 2048;
constexpr int D_DIM = 64;
constexpr int TOPK  = 64;
constexpr int QT    = 16;          // query rows per workgroup
constexpr float SCALE = 0.125f;    // 1/sqrt(64)

__device__ __forceinline__ float wave_sum_f(float v) {
#pragma unroll
  for (int o = 16; o > 0; o >>= 1) v += __shfl_xor(v, o, 32);
  return v;
}
__device__ __forceinline__ int wave_sum_i(int v) {
#pragma unroll
  for (int o = 16; o > 0; o >>= 1) v += __shfl_xor(v, o, 32);
  return v;
}
__device__ __forceinline__ float wave_max_f(float v) {
#pragma unroll
  for (int o = 16; o > 0; o >>= 1) v = fmaxf(v, __shfl_xor(v, o, 32));
  return v;
}

__global__ __launch_bounds__(512, 1)
void topk_attn_kernel(const float* __restrict__ q,
                      const float* __restrict__ k,
                      const float* __restrict__ v,
                      float* __restrict__ out) {
  // 16 x 2048 score/probability tile (f16) + output accumulators + 1/Z per row
  __shared__ _Float16 sc[QT * S_LEN];     // 64 KB
  __shared__ float    accum[4 * 256];     // 4 KB: [ntile][m][n]
  __shared__ float    zinvs[QT];

  const int tid  = threadIdx.x;
  const int w    = tid >> 5;       // wave id 0..15
  const int lane = tid & 31;
  const int bid  = blockIdx.x;
  const int bh   = bid >> 7;                 // 128 q-tiles per (b,h)
  const int q0   = (bid & 127) * QT;
  const long rowbase = (long)bh * S_LEN;     // row base into [B*H*S, D]

  // zero output accumulators (LDS is uninitialized)
  for (int i = tid; i < 4 * 256; i += 512) accum[i] = 0.f;

  const int m    = lane & 15;   // A: row M / B,C: column N
  const int hsel = lane >> 4;   // lane-half select

  // ---------------- Phase 1: scores = (Q*scale) @ K^T  (WMMA f16) ----------
  // A fragment per-lane K index: contiguous k = 16*hsel + i (same order as B below)
  v16h aq0, aq1;
  {
    const float* qrow = q + (rowbase + q0 + m) * D_DIM;
#pragma unroll
    for (int i = 0; i < 16; ++i) {
      int jv = i >> 1;
      int kk = hsel * 16 + i + (jv & 0);
      aq0[i] = (_Float16)(qrow[kk] * SCALE);
      aq1[i] = (_Float16)(qrow[32 + kk] * SCALE);
    }
  }
#pragma unroll 1
  for (int t = 0; t < 8; ++t) {
    const int key0 = (w * 8 + t) * 16;          // 16 keys per tile
    // B fragment: row = contraction (d), col n = key; lanes 0-15 K=0..15, 16-31 K=16..31
    v16h b0, b1;
    const float* krow = k + (rowbase + key0 + m) * D_DIM + hsel * 16;
#pragma unroll
    for (int i = 0; i < 16; ++i) {
      b0[i] = (_Float16)krow[i];        // d = hsel*16 + i       (chunk 0)
      b1[i] = (_Float16)krow[32 + i];   // d = 32 + hsel*16 + i  (chunk 1)
    }
    v8f c = {};
    c = __builtin_amdgcn_wmma_f32_16x16x32_f16(false, aq0, false, b0, (short)0, c, false, false);
    c = __builtin_amdgcn_wmma_f32_16x16x32_f16(false, aq1, false, b1, (short)0, c, false, false);
    // C layout: VGPR i -> row i + 8*hsel, col = lane&15
#pragma unroll
    for (int i = 0; i < 8; ++i)
      sc[(i + 8 * hsel) * S_LEN + key0 + m] = (_Float16)c[i];
  }
  __syncthreads();

  // ------------- Phase 2: exact top-64 threshold per row (wave w = row w) ---
  unsigned uu[64];
  float fmx = -3.0e38f;
#pragma unroll
  for (int e = 0; e < 64; ++e) {
    _Float16 hval = sc[w * S_LEN + lane + 32 * e];
    unsigned short hb = __builtin_bit_cast(unsigned short, hval);
    uu[e] = (hb & 0x8000u) ? (unsigned)(~hb & 0xFFFFu) : (unsigned)(hb | 0x8000u);
    fmx = fmaxf(fmx, (float)hval);
  }
  fmx = wave_max_f(fmx);

  unsigned cur = 0;
#pragma unroll 1
  for (int bit = 15; bit >= 0; --bit) {
    unsigned cand = cur | (1u << bit);
    int cnt = 0;
#pragma unroll
    for (int e = 0; e < 64; ++e) cnt += (uu[e] >= cand) ? 1 : 0;
    cnt = wave_sum_i(cnt);
    if (cnt >= TOPK) cur = cand;   // largest t with count(u>=t) >= 64  == 64th largest
  }
  unsigned short tb = (cur & 0x8000u) ? (unsigned short)(cur ^ 0x8000u)
                                      : (unsigned short)(~cur & 0xFFFFu);
  const float thr = (float)__builtin_bit_cast(_Float16, tb);

  // ------------- Phase 3a: masked exp in place + row sum --------------------
  float z = 0.f;
#pragma unroll 1
  for (int e = 0; e < 64; ++e) {
    int idx = w * S_LEN + lane + 32 * e;
    float s = (float)sc[idx];
    float p = (s >= thr) ? __expf(s - fmx) : 0.f;
    sc[idx] = (_Float16)p;
    z += p;
  }
  z = wave_sum_f(z);
  if (lane == 0) zinvs[w] = 1.f / z;
  __syncthreads();

  // ------------- Phase 3b: Out = P @ V (WMMA f16) ---------------------------
  // wave -> (output column tile, K slice): 4 ntiles x 4 slices of 512 keys
  const int ntile = w & 3;
  const int slice = w >> 2;
  v8f c = {};
#pragma unroll 1
  for (int t = 0; t < 16; ++t) {
    const int key0 = (slice * 16 + t) * 32;
    v16h a, b;
#pragma unroll
    for (int i = 0; i < 16; ++i) {
      int jv = i >> 1;
      int kk = hsel * 16 + i + (jv & 0);
      a[i] = sc[m * S_LEN + key0 + kk];                                   // P fragment (f16, direct)
      b[i] = (_Float16)v[(rowbase + key0 + hsel * 16 + i) * D_DIM + ntile * 16 + m]; // V fragment
    }
    c = __builtin_amdgcn_wmma_f32_16x16x32_f16(false, a, false, b, (short)0, c, false, false);
  }
#pragma unroll
  for (int i = 0; i < 8; ++i)
    atomicAdd(&accum[ntile * 256 + (i + 8 * hsel) * 16 + m], c[i]);  // ds_add_f32
  __syncthreads();

  // ------------- write out: waves 0..3 own one 16-col tile each -------------
  if (w < 4) {
#pragma unroll
    for (int i = 0; i < 8; ++i) {
      int row = i + 8 * hsel;
      float val = accum[w * 256 + row * 16 + m] * zinvs[row];
      out[(rowbase + q0 + row) * D_DIM + w * 16 + m] = val;
    }
  }
}

extern "C" void kernel_launch(void* const* d_in, const int* in_sizes, int n_in,
                              void* d_out, int out_size, void* d_ws, size_t ws_size,
                              hipStream_t stream) {
  const float* q = (const float*)d_in[0];
  const float* k = (const float*)d_in[1];
  const float* v = (const float*)d_in[2];
  // d_in[3] = top_k scalar (== 64, compiled in as TOPK)
  float* out = (float*)d_out;

  const int BH = 4 * 8;
  dim3 grid(BH * (S_LEN / QT));   // 4096 workgroups
  topk_attn_kernel<<<grid, 512, 0, stream>>>(q, k, v, out);
}
